// MoEModel_47296179863618
// MI455X (gfx1250) — matrix operand order repaired and checked by reading, not compiled, not run on hardware
//
#include <hip/hip_runtime.h>
#include <math.h>

#define H 768
#define F 3072
#define NE 8
#define NT 8192
#define TILE_M 32
#define LDA 776   // padded f16 row stride for A tile (bank-conflict free b128)
#define LDH 68    // padded f32 row stride for h slice

typedef __attribute__((ext_vector_type(16))) _Float16 v16h;
typedef __attribute__((ext_vector_type(8)))  _Float16 v8h;
typedef __attribute__((ext_vector_type(4)))  _Float16 v4h;
typedef __attribute__((ext_vector_type(8)))  float    v8f;

union AF { v16h v; v8h h[2]; };

static __device__ __forceinline__ v8f wmma16(v16h a, v16h b, v8f c) {
  return __builtin_amdgcn_wmma_f32_16x16x32_f16(false, a, false, b, (short)0, c, false, false);
}

// ---------------- zero init (out + expert counters) ----------------
__global__ void zero_kernel(float* __restrict__ out, int n, int* __restrict__ cnt) {
  int i = blockIdx.x * blockDim.x + threadIdx.x;
  if (i < n) out[i] = 0.0f;
  if (i < NE) cnt[i] = 0;
}

// ---------------- one-time f32 -> f16 weight conversion ----------------
__global__ void cvt_f32_f16(const float* __restrict__ src, _Float16* __restrict__ dst, int n4) {
  int i = blockIdx.x * blockDim.x + threadIdx.x;
  if (i >= n4) return;
  float4 f = ((const float4*)src)[i];
  v4h h;
  h[0] = (_Float16)f.x; h[1] = (_Float16)f.y;
  h[2] = (_Float16)f.z; h[3] = (_Float16)f.w;
  ((v4h*)dst)[i] = h;
}

// ---------------- gating: softmax + top-2 + bucket build ----------------
__global__ void gate_kernel(const float* __restrict__ x, const float* __restrict__ Wg,
                            float* __restrict__ gateW, int* __restrict__ cnt,
                            int* __restrict__ bucket) {
  int t = blockIdx.x * blockDim.x + threadIdx.x;
  if (t >= NT) return;
  const float* xr = x + (size_t)t * H;
  float lg[NE];
#pragma unroll
  for (int e = 0; e < NE; ++e) lg[e] = 0.0f;
  for (int hh = 0; hh < H; hh += 4) {
    float4 xv = *(const float4*)(xr + hh);
    const float* w0 = Wg + (size_t)hh * NE;
#pragma unroll
    for (int e = 0; e < NE; ++e) {
      lg[e] += xv.x * w0[e] + xv.y * w0[NE + e] + xv.z * w0[2 * NE + e] + xv.w * w0[3 * NE + e];
    }
  }
  float m = lg[0];
#pragma unroll
  for (int e = 1; e < NE; ++e) m = fmaxf(m, lg[e]);
  float p[NE]; float s = 0.0f;
#pragma unroll
  for (int e = 0; e < NE; ++e) { p[e] = expf(lg[e] - m); s += p[e]; }
  float inv = 1.0f / s;
  int i0 = 0;
#pragma unroll
  for (int e = 1; e < NE; ++e) if (p[e] > p[i0]) i0 = e;
  int i1 = (i0 == 0) ? 1 : 0;
#pragma unroll
  for (int e = 0; e < NE; ++e) if (e != i0 && p[e] > p[i1]) i1 = e;
  gateW[t * 2 + 0] = p[i0] * inv;
  gateW[t * 2 + 1] = p[i1] * inv;
  int pos0 = atomicAdd(&cnt[i0], 1);
  bucket[i0 * NT + pos0] = t * 2 + 0;
  int pos1 = atomicAdd(&cnt[i1], 1);
  bucket[i1 * NT + pos1] = t * 2 + 1;
}

// ---------------- fused grouped FFN: x@W1 -> gelu -> @W2 -> scatter ----------------
__global__ __launch_bounds__(256) void moe_ffn_kernel(
    const float* __restrict__ x,
    const _Float16* __restrict__ W1h, const _Float16* __restrict__ W2h,
    const float* __restrict__ b1, const float* __restrict__ b2,
    const float* __restrict__ gateW, const int* __restrict__ cnt,
    const int* __restrict__ bucket, float* __restrict__ out) {
  const int e = blockIdx.y;
  const int n_e = cnt[e];
  const int tile0 = blockIdx.x * TILE_M;
  if (tile0 >= n_e) return;

  __shared__ _Float16 As[TILE_M * LDA];
  __shared__ float    Hs[TILE_M * LDH];
  __shared__ int      rowTok[TILE_M];
  __shared__ float    rowW[TILE_M];

  const int tid = threadIdx.x;
  if (tid < TILE_M) {
    int idx = tile0 + tid;
    if (idx < n_e) {
      int entry = bucket[e * NT + idx];
      rowTok[tid] = entry >> 1;
      rowW[tid] = gateW[entry];
    } else {
      rowTok[tid] = -1;
      rowW[tid] = 0.0f;
    }
  }
  __syncthreads();

  // gather 32 token rows of x, convert f32->f16 into LDS (zeros for pad rows)
  for (int i = tid; i < TILE_M * H; i += 256) {
    int r = i / H, c = i - r * H;
    int tk = rowTok[r];
    float v = (tk >= 0) ? x[(size_t)tk * H + c] : 0.0f;
    As[r * LDA + c] = (_Float16)v;
  }
  __syncthreads();

  const int lane = tid & 31;
  const int wave = tid >> 5;     // 0..7
  const int l15 = lane & 15;
  const int hiH = lane >> 4;     // 0: lanes 0-15, 1: lanes 16-31
  const int mb = wave & 1;       // 16-row block of the 32-token tile
  const int q  = wave >> 1;      // 0..3

  const _Float16* W1e = W1h + (size_t)e * H * F;
  const _Float16* W2e = W2h + (size_t)e * F * H;
  const float* b1e = b1 + (size_t)e * F;
  const float* b2e = b2 + (size_t)e * H;

  v8f acc2[12];
#pragma unroll
  for (int j = 0; j < 12; ++j) {
    v8f z = {0.f, 0.f, 0.f, 0.f, 0.f, 0.f, 0.f, 0.f};
    acc2[j] = z;
  }

  // A-fragment base in LDS: lane m holds K {k0..k0+7, k0+16..23}; lane m+16 holds {k0+8..15, k0+24..31}
  const _Float16* aBase = As + (size_t)(mb * 16 + l15) * LDA + hiH * 8;

  for (int s = 0; s < F / 64; ++s) {
    const int fs = s * 64;

    // ---- FFN1: h tile (mb, nb=q) : rows mb*16.., cols fs+q*16.. ----
    v8f hacc = {0.f, 0.f, 0.f, 0.f, 0.f, 0.f, 0.f, 0.f};
    const _Float16* bBase = W1e + (size_t)(hiH * 8) * F + (fs + q * 16 + l15);  // B: lane = N col, K in table order
#pragma unroll
    for (int kk = 0; kk < H / 32; ++kk) {
      const int k0 = kk * 32;
      AF a, b;
      a.h[0] = *(const v8h*)(aBase + k0);
      a.h[1] = *(const v8h*)(aBase + k0 + 16);
      for (int i = 0; i < 8; ++i) {
        b.v[i] = bBase[(size_t)(k0 + i) * F];
        b.v[8 + i] = bBase[(size_t)(k0 + 16 + i) * F];
      }
      hacc = wmma16(a.v, b.v, hacc);
    }

    // GELU(+b1) in registers, stage h slice (f32, padded) in LDS
    {
      const int hrow = mb * 16 + 8 * hiH;
      const int hcol = q * 16 + l15;
      const float b1v = b1e[fs + hcol];
#pragma unroll
      for (int j = 0; j < 8; ++j) {
        float v = hacc[j] + b1v;
        v = 0.5f * v * (1.0f + erff(v * 0.70710678118654752f));
        Hs[(hrow + j) * LDH + hcol] = v;
      }
    }
    __syncthreads();

    // ---- FFN2: A-fragments from h slice (k local 0..63), reused over 12 N-tiles ----
    AF a0, a1;
    {
      const float* hb = Hs + (size_t)(mb * 16 + l15) * LDH + hiH * 8;
#pragma unroll
      for (int i = 0; i < 8; ++i) {
        a0.h[0][i] = (_Float16)hb[i];
        a0.h[1][i] = (_Float16)hb[16 + i];
        a1.h[0][i] = (_Float16)hb[32 + i];
        a1.h[1][i] = (_Float16)hb[48 + i];
      }
    }
    const _Float16* b2Base = W2e + (size_t)(fs + hiH * 8) * H + l15;  // B: lane = N col, K in table order
#pragma unroll
    for (int j = 0; j < 12; ++j) {
      const int nb2 = q + 4 * j;
      AF b;
      for (int i = 0; i < 8; ++i) {
        b.v[i] = b2Base[(size_t)i * H + nb2 * 16];
        b.v[8 + i] = b2Base[(size_t)(16 + i) * H + nb2 * 16];
      }
      acc2[j] = wmma16(a0.v, b.v, acc2[j]);
      for (int i = 0; i < 8; ++i) {
        b.v[i] = b2Base[(size_t)(32 + i) * H + nb2 * 16];
        b.v[8 + i] = b2Base[(size_t)(48 + i) * H + nb2 * 16];
      }
      acc2[j] = wmma16(a1.v, b.v, acc2[j]);
    }
    __syncthreads();  // protect Hs before next slice overwrites it
  }

  // ---- scale by gate weight (+b2) and scatter-add into out ----
  const int rbase = mb * 16 + 8 * hiH;
#pragma unroll
  for (int j = 0; j < 12; ++j) {
    const int col = (q + 4 * j) * 16 + l15;
    const float b2v = b2e[col];
#pragma unroll
    for (int i = 0; i < 8; ++i) {
      const int r = rbase + i;
      const int tk = rowTok[r];
      if (tk >= 0) {
        unsafeAtomicAdd(out + (size_t)tk * H + col, (acc2[j][i] + b2v) * rowW[r]);
      }
    }
  }
}

extern "C" void kernel_launch(void* const* d_in, const int* in_sizes, int n_in,
                              void* d_out, int out_size, void* d_ws, size_t ws_size,
                              hipStream_t stream) {
  const float* x  = (const float*)d_in[0];
  const float* Wg = (const float*)d_in[1];
  const float* W1 = (const float*)d_in[2];
  const float* b1 = (const float*)d_in[3];
  const float* W2 = (const float*)d_in[4];
  const float* b2 = (const float*)d_in[5];
  float* out = (float*)d_out;

  char* ws = (char*)d_ws;
  size_t off = 0;
  _Float16* W1h = (_Float16*)(ws + off); off += (size_t)NE * H * F * sizeof(_Float16);
  _Float16* W2h = (_Float16*)(ws + off); off += (size_t)NE * F * H * sizeof(_Float16);
  float* gateW  = (float*)(ws + off);    off += (size_t)NT * 2 * sizeof(float);
  int* cnt      = (int*)(ws + off);      off += 256;
  int* bucket   = (int*)(ws + off);      off += (size_t)NE * NT * sizeof(int);

  const int nOut = NT * H;
  zero_kernel<<<(nOut + 255) / 256, 256, 0, stream>>>(out, nOut, cnt);

  const int n4 = NE * H * F / 4;
  cvt_f32_f16<<<(n4 + 255) / 256, 256, 0, stream>>>(W1, W1h, n4);
  cvt_f32_f16<<<(n4 + 255) / 256, 256, 0, stream>>>(W2, W2h, n4);

  gate_kernel<<<NT / 256, 256, 0, stream>>>(x, Wg, gateW, cnt, bucket);

  dim3 grid(NT / TILE_M, NE);
  moe_ffn_kernel<<<grid, 256, 0, stream>>>(x, W1h, W2h, b1, b2, gateW, cnt, bucket, out);
}
